// BlockSparseAttention_13932873908639
// MI455X (gfx1250) — matrix operand order repaired and checked by reading, not compiled, not run on hardware
//
#include <hip/hip_runtime.h>

// CDNA5 / gfx1250: wave32, WMMA 16x16x32 f16->f32
typedef __attribute__((ext_vector_type(16))) _Float16 v16h;
typedef __attribute__((ext_vector_type(8)))  _Float16 v8h;
typedef __attribute__((ext_vector_type(8)))  float    v8f;

#define S_TOT  4096
#define HEADS  16
#define D_DIM  64
#define BLK    64
#define LDP    72   // padded LDS row stride in halves (144B, 16B-aligned)

// ---- WMMA fragment loaders (CDNA5 wave32 VGPR layouts, 16-bit data) ----
// A-matrix 16x32: lane l holds row m=l&15; lanes<16 K={0..7,16..23}, lanes>=16 K={8..15,24..31}
__device__ __forceinline__ v16h load_A_frag(const _Float16* tile, int lane, int kb) {
    const int m   = lane & 15;
    const int off = (lane >> 4) * 8;
    const _Float16* p = tile + m * LDP + kb + off;
    v8h lo = *(const v8h*)(p);
    v8h hi = *(const v8h*)(p + 16);
    v16h a;
#pragma unroll
    for (int i = 0; i < 8; ++i) { a[i] = lo[i]; a[i + 8] = hi[i]; }
    return a;
}

// B-matrix 32x16: lane l holds col n=l&15; lanes<16 K={0..7,16..23}, lanes>=16 K={8..15,24..31}
__device__ __forceinline__ v16h load_B_frag(const _Float16* tile, int lane, int kb) {
    const int n   = lane & 15;
    const int off = (lane >> 4) * 8;
    const _Float16* p = tile + n * LDP + kb + off;
    v8h lo = *(const v8h*)(p);
    v8h hi = *(const v8h*)(p + 16);
    v16h b;
#pragma unroll
    for (int i = 0; i < 8; ++i) { b[i] = lo[i]; b[i + 8] = hi[i]; }
    return b;
}

// ---- Kernel 1: per-head Vsum[h][d] = sum over all S rows of V ----
__global__ __launch_bounds__(256)
void bsattn_vsum_kernel(const float* __restrict__ V, float* __restrict__ vsum_g) {
    __shared__ float part[16][64];
    const int h   = blockIdx.x;
    const int tid = threadIdx.x;
    const int f4  = tid & 15;   // which float4 of the 64-wide row
    const int rg  = tid >> 4;   // row group 0..15
    const float4* v4 = (const float4*)(V + (size_t)h * S_TOT * D_DIM);
    float4 acc = make_float4(0.f, 0.f, 0.f, 0.f);
    for (int r = rg; r < S_TOT; r += 16) {
        float4 v = v4[r * 16 + f4];
        acc.x += v.x; acc.y += v.y; acc.z += v.z; acc.w += v.w;
    }
    part[rg][f4 * 4 + 0] = acc.x;
    part[rg][f4 * 4 + 1] = acc.y;
    part[rg][f4 * 4 + 2] = acc.z;
    part[rg][f4 * 4 + 3] = acc.w;
    __syncthreads();
    if (tid < 64) {
        float s = 0.f;
#pragma unroll
        for (int g = 0; g < 16; ++g) s += part[g][tid];
        vsum_g[h * 64 + tid] = s;
    }
}

// ---- Kernel 2: one WG per (head, diagonal 64x64 block) ----
__global__ __launch_bounds__(256)
void bsattn_block_kernel(const float* __restrict__ Q, const float* __restrict__ K,
                         const float* __restrict__ V, const float* __restrict__ vsum_g,
                         float* __restrict__ Out) {
    __shared__ _Float16 qlds[BLK * LDP];   // Q tile, row-major [q][k]
    __shared__ _Float16 klds[BLK * LDP];   // K tile, row-major [n][k]  (B of QK^T)
    __shared__ _Float16 vlds[BLK * LDP];   // V tile, transposed [d][r] (B of PV)
    __shared__ _Float16 plds[BLK * LDP];   // exp(scores), row-major [q][k]
    __shared__ float zrow[BLK];
    __shared__ float vblk[D_DIM];
    __shared__ float vsum_s[D_DIM];

    const int bq   = blockIdx.x;   // 0..63
    const int h    = blockIdx.y;   // 0..15
    const int tid  = threadIdx.x;
    const int lane = tid & 31;
    const int wave = tid >> 5;     // 8 waves
    const int mi   = wave >> 1;            // output row-tile 0..3
    const int nj   = (wave & 1) * 2;       // output col-tiles nj, nj+1

    const size_t base = ((size_t)h * S_TOT + (size_t)bq * BLK) * D_DIM;

    // ---- stage tiles into LDS (fp32 global -> f16 LDS) ----
    {
        const float4* q4 = (const float4*)(Q + base);
        const float4* k4 = (const float4*)(K + base);
        const float4* v4 = (const float4*)(V + base);
        const int r = tid >> 2;
#pragma unroll
        for (int i = 0; i < 4; ++i) {
            const int f4 = (tid & 3) + 4 * i;
            const int d0 = f4 * 4;
            float4 q = q4[r * 16 + f4];
            float4 k = k4[r * 16 + f4];
            float4 v = v4[r * 16 + f4];
            qlds[r * LDP + d0 + 0] = (_Float16)q.x;
            qlds[r * LDP + d0 + 1] = (_Float16)q.y;
            qlds[r * LDP + d0 + 2] = (_Float16)q.z;
            qlds[r * LDP + d0 + 3] = (_Float16)q.w;
            klds[r * LDP + d0 + 0] = (_Float16)k.x;
            klds[r * LDP + d0 + 1] = (_Float16)k.y;
            klds[r * LDP + d0 + 2] = (_Float16)k.z;
            klds[r * LDP + d0 + 3] = (_Float16)k.w;
            vlds[(d0 + 0) * LDP + r] = (_Float16)v.x;   // transpose V
            vlds[(d0 + 1) * LDP + r] = (_Float16)v.y;
            vlds[(d0 + 2) * LDP + r] = (_Float16)v.z;
            vlds[(d0 + 3) * LDP + r] = (_Float16)v.w;
        }
        if (tid < 64) vsum_s[tid] = vsum_g[h * 64 + tid];
    }
    __syncthreads();

    // ---- GEMM1: S = Q * K^T (scaled), then P = exp(S) into LDS ----
    const float scale = 0.125f;  // 1/sqrt(64)
    v8f acc0 = {}; v8f acc1 = {};
#pragma unroll
    for (int kb = 0; kb < 64; kb += 32) {
        v16h a  = load_A_frag(qlds + mi * 16 * LDP, lane, kb);
        v16h b0 = load_B_frag(klds + (nj + 0) * 16 * LDP, lane, kb);
        v16h b1 = load_B_frag(klds + (nj + 1) * 16 * LDP, lane, kb);
        acc0 = __builtin_amdgcn_wmma_f32_16x16x32_f16(false, a, false, b0,
                                                      (short)0, acc0, false, false);
        acc1 = __builtin_amdgcn_wmma_f32_16x16x32_f16(false, a, false, b1,
                                                      (short)0, acc1, false, false);
    }
    {
        const int col0 = (nj + 0) * 16 + (lane & 15);
        const int col1 = (nj + 1) * 16 + (lane & 15);
        const int rowb = mi * 16 + ((lane >> 4) * 8);
#pragma unroll
        for (int r = 0; r < 8; ++r) {
            plds[(rowb + r) * LDP + col0] = (_Float16)__expf(acc0[r] * scale);
            plds[(rowb + r) * LDP + col1] = (_Float16)__expf(acc1[r] * scale);
        }
    }
    __syncthreads();

    // ---- row sums Z (softmax denom incl. (S-64)*exp(0)) and V column sums ----
    if (tid < 64) {
        float s = 0.f;
#pragma unroll
        for (int k = 0; k < 64; ++k) s += (float)plds[tid * LDP + k];
        zrow[tid] = s + (float)(S_TOT - BLK);
        float c = 0.f;
#pragma unroll
        for (int r = 0; r < 64; ++r) c += (float)vlds[tid * LDP + r];
        vblk[tid] = c;
    }
    __syncthreads();

    // ---- GEMM2: O = P * V ----
    v8f o0 = {}; v8f o1 = {};
#pragma unroll
    for (int kb = 0; kb < 64; kb += 32) {
        v16h a  = load_A_frag(plds + mi * 16 * LDP, lane, kb);
        v16h b0 = load_B_frag(vlds + (nj + 0) * 16 * LDP, lane, kb);
        v16h b1 = load_B_frag(vlds + (nj + 1) * 16 * LDP, lane, kb);
        o0 = __builtin_amdgcn_wmma_f32_16x16x32_f16(false, a, false, b0,
                                                    (short)0, o0, false, false);
        o1 = __builtin_amdgcn_wmma_f32_16x16x32_f16(false, a, false, b1,
                                                    (short)0, o1, false, false);
    }

    // ---- combine: out = (P*V + Vsum - VblockSum) / Z, write to global ----
    {
        const int d0   = (nj + 0) * 16 + (lane & 15);
        const int d1   = (nj + 1) * 16 + (lane & 15);
        const int rowb = mi * 16 + ((lane >> 4) * 8);
        const float corr0 = vsum_s[d0] - vblk[d0];
        const float corr1 = vsum_s[d1] - vblk[d1];
        float* outp = Out + base;
#pragma unroll
        for (int r = 0; r < 8; ++r) {
            const int q = rowb + r;
            const float inv = 1.0f / zrow[q];
            outp[q * 64 + d0] = (o0[r] + corr0) * inv;
            outp[q * 64 + d1] = (o1[r] + corr1) * inv;
        }
    }
}

extern "C" void kernel_launch(void* const* d_in, const int* in_sizes, int n_in,
                              void* d_out, int out_size, void* d_ws, size_t ws_size,
                              hipStream_t stream) {
    (void)in_sizes; (void)n_in; (void)out_size; (void)ws_size;
    const float* Q = (const float*)d_in[0];
    const float* K = (const float*)d_in[1];
    const float* V = (const float*)d_in[2];
    float* Out  = (float*)d_out;
    float* vsum = (float*)d_ws;   // 16*64 floats, fully overwritten each call

    bsattn_vsum_kernel<<<dim3(HEADS), 256, 0, stream>>>(V, vsum);
    bsattn_block_kernel<<<dim3(S_TOT / BLK, HEADS), 256, 0, stream>>>(Q, K, V, vsum, Out);
}
